// SelfAttention_19104014532772
// MI455X (gfx1250) — matrix operand order repaired and checked by reading, not compiled, not run on hardware
//
#include <hip/hip_runtime.h>
#include <hip/hip_bf16.h>

// ---------------------------------------------------------------------------
// Self-attention (q==k shared reduce conv) for B=4, C=256, Cr=64, N=64*64=4096
// fp32 in/out, bf16 WMMA (v_wmma_f32_16x16x32_bf16) for all matmuls.
// Two-pass softmax: pass1 row stats (max, 1/sum), pass2 recompute+PV+residual.
// CDNA5 data movement: TDM tensor_load_to_lds for GEMM tile staging,
// global_load_async_to_lds_b128 for softmax-stat staging.
// ---------------------------------------------------------------------------

typedef __attribute__((ext_vector_type(16))) __bf16 v16bf;
typedef __attribute__((ext_vector_type(8)))  __bf16 v8bf;
typedef __attribute__((ext_vector_type(8)))  float  v8f;
typedef __attribute__((ext_vector_type(4)))  unsigned int v4u;
typedef __attribute__((ext_vector_type(8)))  int v8i;
typedef __attribute__((ext_vector_type(4)))  int v4i;

#define NPIX 4096
#define CFULL 256
#define CRED 64

__device__ inline v8f wmma_bf16(v16bf a, v16bf b, v8f c) {
  return __builtin_amdgcn_wmma_f32_16x16x32_bf16(false, a, false, b,
                                                 (short)0, c, false, false);
}

__device__ inline v8f zero8() {
  v8f z = {0.f, 0.f, 0.f, 0.f, 0.f, 0.f, 0.f, 0.f};
  return z;
}

__device__ inline unsigned lds_off(const void* p) {
  // generic pointer to LDS = {shared aperture, 32-bit LDS offset}
  return (unsigned)(unsigned long long)p;
}

// A-fragment: 16x32 bf16 tile, source row-major [row][k] with row stride ld.
// Lane L (r=L&15, h=L>>4) holds row r, K = {8h..8h+7} U {16+8h..23+8h}.
__device__ inline v16bf load_frag_a(const __bf16* p, int ld) {
  const int lane = threadIdx.x & 31;
  const int r = lane & 15, h = lane >> 4;
  const __bf16* q = p + r * ld + h * 8;
  v8bf lo = *(const v8bf*)(q);
  v8bf hi = *(const v8bf*)(q + 16);
  return __builtin_shufflevector(lo, hi, 0, 1, 2, 3, 4, 5, 6, 7,
                                 8, 9, 10, 11, 12, 13, 14, 15);
}

// B-fragment: 32x16 bf16 tile, source stored K-contiguous per column:
// element (k, n) at p[n*ld + k]. Lane L (n=L&15, h=L>>4) holds col n,
// K = {8h..8h+7} U {16+8h..23+8h}.
__device__ inline v16bf load_frag_b(const __bf16* p, int ld) {
  const int lane = threadIdx.x & 31;
  const int n = lane & 15, h = lane >> 4;
  const __bf16* q = p + n * ld + h * 8;
  v8bf lo = *(const v8bf*)(q);
  v8bf hi = *(const v8bf*)(q + 16);
  return __builtin_shufflevector(lo, hi, 0, 1, 2, 3, 4, 5, 6, 7,
                                 8, 9, 10, 11, 12, 13, 14, 15);
}

// TDM: load a 2D f32 tile [rows x cols], row stride (elements) from global
// into LDS (contiguous rows of `cols` f32). Issue once (caller gates to one
// wave); completion via s_wait_tensorcnt.
__device__ inline void tdm_load_tile_f32(const float* gptr, unsigned ldsaddr,
                                         int cols, int rows, int stride) {
  unsigned long long ga = (unsigned long long)(const void*)gptr;
  v4u g0;
  g0[0] = 1u;                 // count=1, user descriptor
  g0[1] = ldsaddr;            // lds_addr
  g0[2] = (unsigned)ga;       // global_addr[31:0]
  g0[3] = ((unsigned)(ga >> 32) & 0x01FFFFFFu) | 0x80000000u;  // addr[56:32] | type=2
  v8i g1;
  g1[0] = (int)(2u << 16);            // data_size=4B; no multicast/pad/iterate
  g1[1] = (int)((unsigned)cols << 16);  // tensor_dim0[15:0] at bits 63:48
  g1[2] = (int)((unsigned)rows << 16);  // dim0 hi=0 | tensor_dim1[15:0]
  g1[3] = (int)((unsigned)cols << 16);  // dim1 hi=0 | tile_dim0
  g1[4] = rows;                         // tile_dim1 | tile_dim2=0
  g1[5] = stride;                       // tensor_dim0_stride[31:0]
  g1[6] = 0;                            // stride hi | dim1_stride lo
  g1[7] = 0;
  v4i z4 = {0, 0, 0, 0};
  v8i z8 = {0, 0, 0, 0, 0, 0, 0, 0};
  // 6-arg variant (clang-23 / therock-10.0 lane)
  __builtin_amdgcn_tensor_load_to_lds(g0, g1, z4, z4, z8, 0);
}

// ---------------------------------------------------------------------------
// Kernel A: Y = W (MxC f32) * X (CxN f32), written as bf16.
// MODE 0: Y stored transposed [N][M]   (Q^T, M=64)
// MODE 1: Y stored row-major  [M][N]   (V,   M=256)
// Grid: (N/64, B). Block: 256 (8 waves). X tile staged via TDM into LDS,
// then converted/transposed LDS->LDS into the WMMA-B layout.
// ---------------------------------------------------------------------------
template <int M, int MODE>
__global__ __launch_bounds__(256) void qv_gemm(const float* __restrict__ x,
                                               const float* __restrict__ w,
                                               __bf16* __restrict__ dst) {
  constexpr int MT = M / 16;           // m tiles
  constexpr int TILES = MT * 4;        // 4 n-tiles per block
  constexpr int PER_WAVE = TILES / 8;  // tiles per wave
  __shared__ __align__(16) float xraw[32 * 64];  // raw TDM tile [c][n]
  __shared__ __bf16 xT[64 * 40];       // [n_local][c_local], stride 40 (pad)

  const int b = blockIdx.y;
  const int n0 = blockIdx.x * 64;
  const float* xb = x + (size_t)b * CFULL * NPIX;
  const int tid = threadIdx.x;
  const int wv = tid >> 5;
  const int lane = tid & 31;
  const int h = lane >> 4;
  const int jl = lane & 15;

  v8f acc[PER_WAVE];
#pragma unroll
  for (int i = 0; i < PER_WAVE; ++i) acc[i] = zero8();

  for (int c0 = 0; c0 < CFULL; c0 += 32) {
    // --- TDM: stage X[c0..c0+31][n0..n0+63] f32 into LDS (wave 0 issues) ---
    if (wv == 0) {
      tdm_load_tile_f32(xb + (size_t)c0 * NPIX + n0, lds_off(&xraw[0]),
                        /*cols=*/64, /*rows=*/32, /*stride=*/NPIX);
      __builtin_amdgcn_s_wait_tensorcnt(0);
    }
    __syncthreads();

    // --- convert + transpose LDS->LDS into bf16 WMMA-B layout ---
    {
      const int cl = tid >> 3;          // 0..31
      const int nl = (tid & 7) * 8;     // 0..56 step 8
      const float4 f0 = *(const float4*)&xraw[cl * 64 + nl];
      const float4 f1 = *(const float4*)&xraw[cl * 64 + nl + 4];
      xT[(nl + 0) * 40 + cl] = (__bf16)f0.x;
      xT[(nl + 1) * 40 + cl] = (__bf16)f0.y;
      xT[(nl + 2) * 40 + cl] = (__bf16)f0.z;
      xT[(nl + 3) * 40 + cl] = (__bf16)f0.w;
      xT[(nl + 4) * 40 + cl] = (__bf16)f1.x;
      xT[(nl + 5) * 40 + cl] = (__bf16)f1.y;
      xT[(nl + 6) * 40 + cl] = (__bf16)f1.z;
      xT[(nl + 7) * 40 + cl] = (__bf16)f1.w;
    }
    __syncthreads();

#pragma unroll
    for (int i = 0; i < PER_WAVE; ++i) {
      const int t = wv + 8 * i;
      const int mt = t % MT;
      const int nt = t / MT;
      // A fragment from W (f32 -> bf16 on the fly)
      const float* wp = w + (size_t)(mt * 16) * CFULL + c0;
      const float* wq = wp + (lane & 15) * CFULL + h * 8;
      v16bf af;
#pragma unroll
      for (int e = 0; e < 8; ++e) {
        af[e] = (__bf16)wq[e];
        af[e + 8] = (__bf16)wq[e + 16];
      }
      v16bf bf = load_frag_b(&xT[nt * 16 * 40], 40);
      acc[i] = wmma_bf16(af, bf, acc[i]);
    }
    __syncthreads();
  }

  // --- store bf16 result ---
#pragma unroll
  for (int i = 0; i < PER_WAVE; ++i) {
    const int t = wv + 8 * i;
    const int mt = t % MT;
    const int nt = t / MT;
    const int ncol = n0 + nt * 16 + jl;
#pragma unroll
    for (int v = 0; v < 8; ++v) {
      const int mrow = mt * 16 + v + 8 * h;
      if (MODE == 0) {  // Q^T: [N][M]
        dst[(size_t)b * NPIX * M + (size_t)ncol * M + mrow] = (__bf16)acc[i][v];
      } else {          // V: [M][N]
        dst[(size_t)b * M * NPIX + (size_t)mrow * NPIX + ncol] = (__bf16)acc[i][v];
      }
    }
  }
}

// ---------------------------------------------------------------------------
// Kernel B: row softmax stats. att[n,m] = sum_c qT[n][c]*qT[m][c].
// Each block: one batch, 16 rows n. 8 waves stripe over m tiles, online
// max/sum per lane element, then serial LDS reduction (one thread per row).
// Outputs rmax[n] and 1/sum (reciprocal, so pass2 multiplies).
// ---------------------------------------------------------------------------
__global__ __launch_bounds__(256) void row_stats(const __bf16* __restrict__ qT,
                                                 float* __restrict__ rmax,
                                                 float* __restrict__ rsuminv) {
  __shared__ float redM[16 * 128];
  __shared__ float redS[16 * 128];
  const int b = blockIdx.y;
  const int n0 = blockIdx.x * 16;
  const __bf16* qb = qT + (size_t)b * NPIX * CRED;
  const int tid = threadIdx.x;
  const int wv = tid >> 5;
  const int lane = tid & 31;
  const int h = lane >> 4;

  const v16bf a0 = load_frag_a(qb + (size_t)n0 * CRED, CRED);
  const v16bf a1 = load_frag_a(qb + (size_t)n0 * CRED + 32, CRED);

  float mx[8], sm[8];
#pragma unroll
  for (int v = 0; v < 8; ++v) { mx[v] = -3.0e38f; sm[v] = 0.f; }

  for (int m0 = wv * 16; m0 < NPIX; m0 += 128) {
    const __bf16* bp = qb + (size_t)m0 * CRED;
    if (m0 + 128 < NPIX) __builtin_prefetch(bp + 128 * CRED, 0, 0);
    v16bf b0 = load_frag_b(bp, CRED);
    v16bf b1 = load_frag_b(bp + 32, CRED);
    v8f s = zero8();
    s = wmma_bf16(a0, b0, s);
    s = wmma_bf16(a1, b1, s);
#pragma unroll
    for (int v = 0; v < 8; ++v) {
      const float val = s[v];
      const float nm = fmaxf(mx[v], val);
      sm[v] = sm[v] * __expf(mx[v] - nm) + __expf(val - nm);
      mx[v] = nm;
    }
  }

  const int pc = wv * 16 + (lane & 15);
#pragma unroll
  for (int v = 0; v < 8; ++v) {
    const int row = v + 8 * h;
    redM[row * 128 + pc] = mx[v];
    redS[row * 128 + pc] = sm[v];
  }
  __syncthreads();
  if (tid < 16) {
    float M = -3.0e38f;
    for (int j = 0; j < 128; ++j) M = fmaxf(M, redM[tid * 128 + j]);
    float S = 0.f;
    for (int j = 0; j < 128; ++j)
      S += redS[tid * 128 + j] * __expf(redM[tid * 128 + j] - M);
    rmax[(size_t)b * NPIX + n0 + tid] = M;
    rsuminv[(size_t)b * NPIX + n0 + tid] = 1.0f / S;
  }
}

// ---------------------------------------------------------------------------
// Kernel C: out[c, j] = x[c, j] + sum_n V[c, n] * beta[n, j].
// Block: one batch, 32 output columns (2 j-tiles). Waves 0..3 compute the
// 32x32 logit block (2 n-halves x 2 j-tiles), exp with precomputed stats,
// stage P^T in LDS; all 8 waves then do the P*V WMMAs (16 c-tiles x 2 j).
// Softmax stats staged via global_load_async_to_lds_b128 (ASYNCcnt).
// ---------------------------------------------------------------------------
__global__ __launch_bounds__(256) void attn_apply(
    const __bf16* __restrict__ qT, const __bf16* __restrict__ V,
    const float* __restrict__ x, const float* __restrict__ rmax,
    const float* __restrict__ rsuminv, float* __restrict__ out) {
  __shared__ __bf16 pT[2][16 * 40];  // per j-tile: [j_local][n_local(32)], pad 40
  __shared__ __align__(16) float stats[64];  // [0..31]=rowmax, [32..63]=1/sum

  const int b = blockIdx.y;
  const int j0 = blockIdx.x * 32;
  const __bf16* qb = qT + (size_t)b * NPIX * CRED;
  const __bf16* vb = V + (size_t)b * CFULL * NPIX;
  const float* xb = x + (size_t)b * CFULL * NPIX;
  float* ob = out + (size_t)b * CFULL * NPIX;
  const float* rmx = rmax + (size_t)b * NPIX;
  const float* rsi = rsuminv + (size_t)b * NPIX;

  const int tid = threadIdx.x;
  const int wv = tid >> 5;
  const int lane = tid & 31;
  const int h = lane >> 4;
  const int jl = lane & 15;

  // Waves 0..3 own one (n_half, j_tile) logit tile each.
  const int jt_s = wv >> 1;
  const int nh_s = wv & 1;
  v16bf bj0 = {}, bj1 = {};
  if (wv < 4) {
    const __bf16* p = qb + (size_t)(j0 + jt_s * 16) * CRED;
    bj0 = load_frag_b(p, CRED);
    bj1 = load_frag_b(p + 32, CRED);
  }

  v8f accPV[4];
#pragma unroll
  for (int i = 0; i < 4; ++i) accPV[i] = zero8();

  for (int n0 = 0; n0 < NPIX; n0 += 32) {
    // Async-stage 32 rowmax + 32 rsuminv f32 (16 lanes x 16B) into LDS.
    if (tid < 16) {
      const float* gsrc = (tid < 8) ? (rmx + n0 + (tid << 2))
                                    : (rsi + n0 + ((tid - 8) << 2));
      unsigned ldst = lds_off(&stats[tid * 4]);
      asm volatile("global_load_async_to_lds_b128 %0, %1, off"
                   :: "v"(ldst), "v"(gsrc) : "memory");
    }
    v8f s = zero8();
    if (wv < 4) {
      const __bf16* ap = qb + (size_t)(n0 + nh_s * 16) * CRED;
      v16bf a0 = load_frag_a(ap, CRED);
      v16bf a1 = load_frag_a(ap + 32, CRED);
      s = wmma_bf16(a0, bj0, s);
      s = wmma_bf16(a1, bj1, s);
    }
    if (wv == 0) asm volatile("s_wait_asynccnt 0x0" ::: "memory");
    __syncthreads();  // stats landed; prior-iter PV reads of pT done
    if (wv < 4) {
#pragma unroll
      for (int v = 0; v < 8; ++v) {
        const int nr = nh_s * 16 + v + 8 * h;  // n index within [0,32)
        const float p = __expf(s[v] - stats[nr]) * stats[32 + nr];
        pT[jt_s][jl * 40 + nr] = (__bf16)p;
      }
    }
    __syncthreads();  // P staged
#pragma unroll
    for (int i = 0; i < 4; ++i) {
      const int tile = wv + i * 8;   // 32 tiles over 8 waves
      const int ct = tile & 15;
      const int jt = tile >> 4;
      const __bf16* vp = vb + (size_t)(ct * 16) * NPIX + n0;
      if (n0 + 32 < NPIX) __builtin_prefetch(vp + 32, 0, 0);
      v16bf av = load_frag_a(vp, NPIX);
      v16bf bp = load_frag_b(&pT[jt][0], 40);
      accPV[i] = wmma_bf16(av, bp, accPV[i]);
    }
  }

  // Residual + store (f32)
#pragma unroll
  for (int i = 0; i < 4; ++i) {
    const int tile = wv + i * 8;
    const int ct = tile & 15;
    const int jt = tile >> 4;
    const int col = j0 + jt * 16 + jl;
#pragma unroll
    for (int v = 0; v < 8; ++v) {
      const int row = ct * 16 + v + 8 * h;
      const size_t idx = (size_t)row * NPIX + col;
      ob[idx] = xb[idx] + accPV[i][v];
    }
  }
}

// ---------------------------------------------------------------------------
extern "C" void kernel_launch(void* const* d_in, const int* in_sizes, int n_in,
                              void* d_out, int out_size, void* d_ws,
                              size_t ws_size, hipStream_t stream) {
  const float* x = (const float*)d_in[0];         // [4,256,64,64]
  const float* w_reduce = (const float*)d_in[1];  // [64,256]
  const float* w_full = (const float*)d_in[2];    // [256,256]
  float* out = (float*)d_out;                     // [4,256,64,64]

  char* ws = (char*)d_ws;
  __bf16* qT = (__bf16*)(ws);                          // 4*4096*64  bf16 = 2 MB
  __bf16* Vm = (__bf16*)(ws + (2u << 20));             // 4*256*4096 bf16 = 8 MB
  float* rmax = (float*)(ws + (10u << 20));            // 4*4096 f32 = 64 KB
  float* rsuminv = (float*)(ws + (10u << 20) + (64u << 10));

  qv_gemm<64, 0><<<dim3(64, 4), 256, 0, stream>>>(x, w_reduce, qT);
  qv_gemm<256, 1><<<dim3(64, 4), 256, 0, stream>>>(x, w_full, Vm);
  row_stats<<<dim3(256, 4), 256, 0, stream>>>(qT, rmax, rsuminv);
  attn_apply<<<dim3(128, 4), 256, 0, stream>>>(qT, Vm, x, rmax, rsuminv, out);
}
